// RNN_88261577933173
// MI455X (gfx1250) — matrix operand order repaired and checked by reading, not compiled, not run on hardware
//
#include <hip/hip_runtime.h>

// Problem constants (reference: B=64, T=512, E=256, H=128)
#define BN 64
#define TN 512
#define EN 256
#define HN 128
#define G4 512  // 4*H

typedef __attribute__((ext_vector_type(16))) __bf16 v16bf;
typedef __attribute__((ext_vector_type(8)))  float  v8f;

union Frag16 { v16bf v; unsigned int u[8]; };

__device__ __forceinline__ unsigned short f2bf(float f) {
  unsigned int u = __float_as_uint(f);
  u += 0x7FFFu + ((u >> 16) & 1u);   // round-to-nearest-even
  return (unsigned short)(u >> 16);
}

__device__ __forceinline__ float sigm(float x) { return 1.0f / (1.0f + __expf(-x)); }

// ---------------------------------------------------------------- prep kernels
__global__ void k_f32_to_bf16(const float* __restrict__ src,
                              unsigned short* __restrict__ dst, int n) {
  int i = blockIdx.x * blockDim.x + threadIdx.x;
  if (i < n) dst[i] = f2bf(src[i]);
}

__global__ void k_bias_sum(const float* __restrict__ a, const float* __restrict__ b,
                           float* __restrict__ o, int n) {
  int i = blockIdx.x * blockDim.x + threadIdx.x;
  if (i < n) o[i] = a[i] + b[i];
}

// Row softmax over width 128; one wave32 per row (4 elements per lane).
__global__ void k_row_softmax128(const float* __restrict__ src,
                                 float* __restrict__ dst, int rows) {
  int row = blockIdx.x * 8 + (threadIdx.x >> 5);
  if (row >= rows) return;
  int lane = threadIdx.x & 31;
  const float* p = src + (long)row * HN;
  float v[4];
  float m = -3.4e38f;
#pragma unroll
  for (int j = 0; j < 4; ++j) { v[j] = p[lane + 32 * j]; m = fmaxf(m, v[j]); }
#pragma unroll
  for (int off = 16; off > 0; off >>= 1) m = fmaxf(m, __shfl_xor(m, off, 32));
  float s = 0.f;
#pragma unroll
  for (int j = 0; j < 4; ++j) { v[j] = __expf(v[j] - m); s += v[j]; }
#pragma unroll
  for (int off = 16; off > 0; off >>= 1) s += __shfl_xor(s, off, 32);
  float inv = 1.0f / s;
  float* q = dst + (long)row * HN;
#pragma unroll
  for (int j = 0; j < 4; ++j) q[lane + 32 * j] = v[j] * inv;
}

// ------------------------------------------------- input projection (big GEMM)
// Xp[m, n] = sum_k Abf[m, k] * Wih[n, k] + bias[n]
// M = B*T = 32768, N = 4H = 512, K = E = 256. bf16 WMMA, f32 accumulate.
__global__ void k_gemm_in(const unsigned short* __restrict__ Abf,
                          const unsigned short* __restrict__ Wih,
                          const float* __restrict__ bias,
                          float* __restrict__ Xp) {
  const int lane = threadIdx.x & 31;
  const int wave = threadIdx.x >> 5;          // 0..7
  const int mt = blockIdx.x * 8 + wave;       // 0..2047
  const int nt = blockIdx.y;                  // 0..31
  const int hi = (lane >> 4) & 1;
  const int l15 = lane & 15;

  const int arow = mt * 16 + l15;
  const int bcol = nt * 16 + l15;

  v8f acc = {};
#pragma unroll
  for (int kk = 0; kk < EN / 32; ++kk) {
    Frag16 a, b;
#pragma unroll
    for (int v = 0; v < 8; ++v) {
      // A (16x32 bf16): lanes0-15 K={0..7,16..23}, lanes16-31 +8; pairs per dword
      int ka = kk * 32 + hi * 8 + ((v & 4) ? 16 : 0) + 2 * (v & 3);
      a.u[v] = *(const unsigned int*)(Abf + (long)arow * EN + ka);
      // B (32x16 bf16): lanes0-15 K={0..7,16..23}, lanes16-31 +8; pairs per dword
      int kb = kk * 32 + hi * 8 + ((v & 4) ? 16 : 0) + 2 * (v & 3);
      b.u[v] = *(const unsigned int*)(Wih + (long)bcol * EN + kb);
    }
    acc = __builtin_amdgcn_wmma_f32_16x16x32_bf16(false, a.v, false, b.v,
                                                  (short)0, acc, false, false);
  }
  const int n = nt * 16 + l15;
  const float bn = bias[n];
#pragma unroll
  for (int r = 0; r < 8; ++r) {
    int m = mt * 16 + hi * 8 + r;  // C/D layout: VGPR r -> row r (+8 for hi lanes)
    Xp[(long)m * G4 + n] = acc[r] + bn;
  }
}

// ------------------------------------------------------------- LSTM recurrence
// 4 workgroups x 512 threads (16 waves). Each workgroup owns 16 batch rows.
// W_hh (bf16), h (bf16), c (f32), gate staging all live in LDS (172 KB < 320 KB).
// Xp tiles for step t+1 are prefetched (global_prefetch_b8) during step t to
// hide the only global-memory latency on the serial critical path.
__global__ void k_lstm(const unsigned short* __restrict__ Whh,
                       const float* __restrict__ Xp,
                       float* __restrict__ hs) {
  extern __shared__ char smem[];
  unsigned short* whh = (unsigned short*)smem;             // 512*128 bf16 = 128 KB
  unsigned short* hbf = (unsigned short*)(smem + 131072);  // 16*128 bf16 = 4 KB
  float* cArr = (float*)(smem + 135168);                   // 16*128 f32  = 8 KB
  float* Gs   = (float*)(smem + 143360);                   // 16*512 f32  = 32 KB

  const int tid = threadIdx.x;        // 0..511
  const int lane = tid & 31;
  const int wave = tid >> 5;          // 0..15
  const int hi = (lane >> 4) & 1;
  const int l15 = lane & 15;
  const int bbase = blockIdx.x * 16;

  // Stage W_hh into LDS (32768 dwords), zero h and c.
  {
    const unsigned int* src = (const unsigned int*)Whh;
    unsigned int* dst = (unsigned int*)whh;
    for (int i = tid; i < (G4 * HN) / 2; i += 512) dst[i] = src[i];
    for (int i = tid; i < 16 * HN; i += 512) { hbf[i] = 0; cArr[i] = 0.0f; }
  }
  __syncthreads();

  for (int t = 0; t < TN; ++t) {
    // G = Xp_t + h @ Whh^T : each wave computes 2 of the 32 N-tiles.
#pragma unroll
    for (int tt = 0; tt < 2; ++tt) {
      const int nt = wave * 2 + tt;         // 0..31
      const int n = nt * 16 + l15;
      v8f acc;
#pragma unroll
      for (int r = 0; r < 8; ++r) {
        int b = bbase + hi * 8 + r;
        acc[r] = Xp[((long)b * TN + t) * G4 + n];  // C operand preload
      }
      // Prefetch next step's Xp tile rows into near caches (fire-and-forget).
      if (t + 1 < TN) {
#pragma unroll
        for (int r = 0; r < 8; ++r) {
          int b = bbase + hi * 8 + r;
          __builtin_prefetch(&Xp[((long)b * TN + (t + 1)) * G4 + n], 0, 3);
        }
      }
#pragma unroll
      for (int kk = 0; kk < HN / 32; ++kk) {
        Frag16 a, bf;
#pragma unroll
        for (int v = 0; v < 8; ++v) {
          int ka = kk * 32 + hi * 8 + ((v & 4) ? 16 : 0) + 2 * (v & 3);
          a.u[v] = *(const unsigned int*)(hbf + l15 * HN + ka);
          int kb = kk * 32 + hi * 8 + ((v & 4) ? 16 : 0) + 2 * (v & 3);
          bf.u[v] = *(const unsigned int*)(whh + n * HN + kb);
        }
        acc = __builtin_amdgcn_wmma_f32_16x16x32_bf16(false, a.v, false, bf.v,
                                                      (short)0, acc, false, false);
      }
#pragma unroll
      for (int r = 0; r < 8; ++r) Gs[(hi * 8 + r) * G4 + n] = acc[r];
    }
    __syncthreads();

    // Fused gates: 16*128 = 2048 (h,c) elements over 512 threads (4 each).
    {
      const int m = tid >> 5;             // 0..15
      const int n0 = (tid & 31) * 4;
      const int b = bbase + m;
#pragma unroll
      for (int j = 0; j < 4; ++j) {
        int n = n0 + j;
        float gi = Gs[m * G4 + n];
        float gf = Gs[m * G4 + HN + n];
        float gg = Gs[m * G4 + 2 * HN + n];
        float go = Gs[m * G4 + 3 * HN + n];
        float cv = cArr[m * HN + n];
        cv = sigm(gf) * cv + sigm(gi) * tanhf(gg);
        float hv = sigm(go) * tanhf(cv);
        cArr[m * HN + n] = cv;
        hbf[m * HN + n] = f2bf(hv);
        hs[((long)b * TN + t) * HN + n] = hv;
      }
    }
    __syncthreads();
  }
}

// --------------------------------------------------------------- CRF forward
// One 512-thread block per batch element. Thread (j = tid&127, part = tid>>7)
// reduces i in [32*part, 32*part+32); part-0 waves merge the 4 partial
// (max,sum) pairs with a log-add combine. trL has stride-129 padding so the
// column walk is LDS-bank-conflict-free; cur[i] reads are wave broadcasts.
__global__ void k_crf(const float* __restrict__ Hs, const int* __restrict__ labels,
                      const float* __restrict__ transS, float* __restrict__ scoreB) {
  extern __shared__ char smem[];
  float* trL = (float*)smem;               // 128*129 f32 = 66048 B
  float* red = (float*)(smem + 66048);     // 512 f32    =  2048 B
  float* pA  = (float*)(smem + 68096);     // 128 f32    =   512 B
  float* pB  = (float*)(smem + 68608);     // 128 f32    =   512 B
  float* pm  = (float*)(smem + 69120);     // 4*128 f32  =  2048 B
  float* ps  = (float*)(smem + 71168);     // 4*128 f32  =  2048 B (end 73216)
  const int tid = threadIdx.x;             // 0..511
  const int j = tid & 127;
  const int part = tid >> 7;               // 0..3 (wave-uniform)
  const int b = blockIdx.x;

  for (int idx = tid; idx < HN * HN; idx += 512)
    trL[(idx >> 7) * 129 + (idx & 127)] = transS[idx];
  __syncthreads();

  // Gold-path score Pr = sum_t emit[b,t,l_t] + sum_t trans[l_t, l_{t+1}]
  float pr = 0.f;
  for (int t = tid; t < TN; t += 512) {
    int l = labels[b * TN + t];
    pr += Hs[((long)b * TN + t) * HN + l];
    if (t + 1 < TN) pr += trL[l * 129 + labels[b * TN + t + 1]];
  }
  red[tid] = pr;
  __syncthreads();
  for (int s = 256; s > 0; s >>= 1) { if (tid < s) red[tid] += red[tid + s]; __syncthreads(); }
  const float Pr = red[0];
  __syncthreads();

  // Forward algorithm: pre0 = H[:,0,:] + trans[0]
  float* cur = pA;
  float* nxt = pB;
  if (part == 0) cur[j] = Hs[((long)b * TN) * HN + j] + trL[j];
  __syncthreads();
  const int i0 = part * 32;
  for (int t = 1; t < TN; ++t) {
    float m = -3.4e38f;
    for (int i = i0; i < i0 + 32; ++i) m = fmaxf(m, cur[i] + trL[i * 129 + j]);
    float s = 0.f;
    for (int i = i0; i < i0 + 32; ++i) s += __expf(cur[i] + trL[i * 129 + j] - m);
    pm[part * 128 + j] = m;
    ps[part * 128 + j] = s;
    __syncthreads();
    if (part == 0) {
      float m0 = pm[j], m1 = pm[128 + j], m2 = pm[256 + j], m3 = pm[384 + j];
      float M = fmaxf(fmaxf(m0, m1), fmaxf(m2, m3));
      float S = ps[j] * __expf(m0 - M) + ps[128 + j] * __expf(m1 - M) +
                ps[256 + j] * __expf(m2 - M) + ps[384 + j] * __expf(m3 - M);
      nxt[j] = Hs[((long)b * TN + t) * HN + j] + M + __logf(S);
    }
    __syncthreads();
    float* tp = cur; cur = nxt; nxt = tp;
  }

  // Ps = logsumexp(pre + trans[H-1])
  if (part == 0) red[j] = cur[j] + trL[127 * 129 + j];
  __syncthreads();
  for (int s = 64; s > 0; s >>= 1) { if (tid < s) red[tid] = fmaxf(red[tid], red[tid + s]); __syncthreads(); }
  float M = red[0];
  __syncthreads();
  if (part == 0) red[j] = __expf(cur[j] + trL[127 * 129 + j] - M);
  __syncthreads();
  for (int s = 64; s > 0; s >>= 1) { if (tid < s) red[tid] += red[tid + s]; __syncthreads(); }
  if (tid == 0) scoreB[b] = (M + __logf(red[0])) - Pr;
}

// Deterministic fixed-order final reduction; also (re)initializes d_out.
__global__ void k_final(const float* __restrict__ scoreB, float* __restrict__ out) {
  __shared__ float red[64];
  int tid = threadIdx.x;
  red[tid] = scoreB[tid];
  __syncthreads();
  for (int s = 32; s > 0; s >>= 1) { if (tid < s) red[tid] += red[tid + s]; __syncthreads(); }
  if (tid == 0) out[0] = red[0];
}

// ------------------------------------------------------------------ launcher
extern "C" void kernel_launch(void* const* d_in, const int* in_sizes, int n_in,
                              void* d_out, int out_size, void* d_ws, size_t ws_size,
                              hipStream_t stream) {
  (void)in_sizes; (void)n_in; (void)out_size; (void)ws_size;
  const float* inputs     = (const float*)d_in[0];  // (B,T,E)
  const int*   labels     = (const int*)d_in[1];    // (B,T)
  const float* W_ih       = (const float*)d_in[2];  // (4H,E)
  const float* W_hh       = (const float*)d_in[3];  // (4H,H)
  const float* b_ih       = (const float*)d_in[4];  // (4H)
  const float* b_hh       = (const float*)d_in[5];  // (4H)
  const float* transition = (const float*)d_in[6];  // (H,H)
  float* out = (float*)d_out;

  // Workspace layout (bytes)
  char* ws = (char*)d_ws;
  float*          Xp     = (float*)(ws);                              // 64 MB
  unsigned short* Abf    = (unsigned short*)(ws + (size_t)67108864);  // 16 MB
  float*          hs     = (float*)(ws + (size_t)83886080);           // 16 MB
  unsigned short* Wih    = (unsigned short*)(ws + (size_t)100663296); // 256 KB
  unsigned short* Whh    = (unsigned short*)(ws + (size_t)100925440); // 128 KB
  float*          bias   = (float*)(ws + (size_t)101056512);          // 2 KB
  float*          transS = (float*)(ws + (size_t)101058560);          // 64 KB
  float*          scoreB = (float*)(ws + (size_t)101124096);          // 256 B

  const int nIn = BN * TN * EN;   // 8388608
  const int nWih = G4 * EN;       // 131072
  const int nWhh = G4 * HN;       // 65536

  k_f32_to_bf16<<<dim3((nIn + 255) / 256), dim3(256), 0, stream>>>(inputs, Abf, nIn);
  k_f32_to_bf16<<<dim3((nWih + 255) / 256), dim3(256), 0, stream>>>(W_ih, Wih, nWih);
  k_f32_to_bf16<<<dim3((nWhh + 255) / 256), dim3(256), 0, stream>>>(W_hh, Whh, nWhh);
  k_bias_sum<<<dim3(2), dim3(256), 0, stream>>>(b_ih, b_hh, bias, G4);
  k_row_softmax128<<<dim3(16), dim3(256), 0, stream>>>(transition, transS, HN);

  // Input projection GEMM: M-tiles = 2048 (8 per block), N-tiles = 32.
  k_gemm_in<<<dim3(256, 32), dim3(256), 0, stream>>>(Abf, Wih, bias, Xp);

  // LSTM recurrence: 4 workgroups x 512 threads, 176128 B dynamic LDS.
  k_lstm<<<dim3(4), dim3(512), 176128, stream>>>(Whh, Xp, hs);

  // Softmax over H, in place (32768 rows, 8 rows per 256-thread block).
  k_row_softmax128<<<dim3(4096), dim3(256), 0, stream>>>(hs, hs, BN * TN);

  // CRF forward + gold-path score: 64 blocks x 512 threads, 73216 B LDS.
  k_crf<<<dim3(64), dim3(512), 73216, stream>>>(hs, labels, transS, scoreB);

  k_final<<<dim3(1), dim3(64), 0, stream>>>(scoreB, out);
}
